// NodeSetAttention_3796751090098
// MI455X (gfx1250) — matrix operand order repaired and checked by reading, not compiled, not run on hardware
//
#include <hip/hip_runtime.h>
#include <hip/hip_bf16.h>
#include <math.h>

// ---------------------------------------------------------------------------
// Problem constants (match reference)
// ---------------------------------------------------------------------------
#define BB 4
#define SS 2048
#define DD 256
#define HH 8
#define HDIM 32
#define DFF 1024          // 4*D
#define NROWS (BB * SS)   // 8192

typedef __attribute__((ext_vector_type(16))) _Float16 v16h;
typedef __attribute__((ext_vector_type(8)))  float    v8f;
typedef __attribute__((ext_vector_type(4)))  int      v4i;

union Frag16 {
    v16h v;
    uint4 u[2];
    _Float16 h[16];
};

static __device__ __forceinline__ v8f wmma_f16(const Frag16& a, const Frag16& b, v8f c) {
    return __builtin_amdgcn_wmma_f32_16x16x32_f16(false, a.v, false, b.v,
                                                  (short)0, c, false, false);
}

// ---------------------------------------------------------------------------
// Async global->LDS staging (gfx1250 GLOBAL_LOAD_ASYNC_TO_LDS_B128, ASYNCcnt).
// Builtin signature (from hipcc diagnostic): (AS1 v4i* src, AS3 v4i* lds, imm, imm).
// Per CDNA5 ISA 10.2: generic LDS address low 32 bits == LDS byte offset, so
// an integer round-trip legally produces the AS3 pointer.
// Falls back to plain LDS stores when the builtin is not declared.
// ---------------------------------------------------------------------------
#ifdef __has_builtin
#if __has_builtin(__builtin_amdgcn_global_load_async_to_lds_b128) && \
    __has_builtin(__builtin_amdgcn_s_wait_asynccnt)
#define HAVE_ASYNC_LDS 1
#endif
#endif
#ifndef HAVE_ASYNC_LDS
#define HAVE_ASYNC_LDS 0
#endif

static __device__ __forceinline__ void copy16_g2l(_Float16* dst, const _Float16* src) {
#if HAVE_ASYNC_LDS
    __attribute__((address_space(1))) v4i* gp =
        (__attribute__((address_space(1))) v4i*)(unsigned long long)src;
    __attribute__((address_space(3))) v4i* lp =
        (__attribute__((address_space(3))) v4i*)(unsigned)(unsigned long long)dst;
    __builtin_amdgcn_global_load_async_to_lds_b128(gp, lp, 0, 0);
#else
    *(uint4*)dst = *(const uint4*)src;
#endif
}

static __device__ __forceinline__ void async_copy_wait() {
#if HAVE_ASYNC_LDS
    __builtin_amdgcn_s_wait_asynccnt(0);
#endif
}

// ---------------------------------------------------------------------------
// Utility kernels
// ---------------------------------------------------------------------------
__global__ void fill_zero_f32(float* p, long long n) {
    long long i = (long long)blockIdx.x * blockDim.x + threadIdx.x;
    if (i < n) p[i] = 0.0f;
}

__global__ void cvt_f32_to_f16(const float* __restrict__ in, _Float16* __restrict__ out, int n) {
    int i = blockIdx.x * blockDim.x + threadIdx.x;
    if (i < n) out[i] = (_Float16)in[i];
}

// scatter clipped edge_attr into bias2d[S][S]
__global__ void scatter_bias_kernel(const float* __restrict__ edge_attr,
                                    const int* __restrict__ edge_index,
                                    float* __restrict__ bias2d, int E) {
    int i = blockIdx.x * blockDim.x + threadIdx.x;
    if (i >= E) return;
    int r = edge_index[i];
    int c = edge_index[E + i];
    float v = fminf(fmaxf(edge_attr[i], -5.0f), 5.0f);
    bias2d[(size_t)r * SS + c] = v;
}

// LayerNorm over 256 columns, output f16. one block (256 threads) per row.
__global__ void ln_kernel(const float* __restrict__ x, const float* __restrict__ g,
                          const float* __restrict__ b, _Float16* __restrict__ out) {
    __shared__ float red[DD];
    int row = blockIdx.x;
    int tid = threadIdx.x;
    float v = x[(size_t)row * DD + tid];
    red[tid] = v;
    __syncthreads();
    for (int s = DD / 2; s > 0; s >>= 1) {
        if (tid < s) red[tid] += red[tid + s];
        __syncthreads();
    }
    float mu = red[0] * (1.0f / DD);
    __syncthreads();
    float d = v - mu;
    red[tid] = d * d;
    __syncthreads();
    for (int s = DD / 2; s > 0; s >>= 1) {
        if (tid < s) red[tid] += red[tid + s];
        __syncthreads();
    }
    float var = red[0] * (1.0f / DD);
    float y = d * rsqrtf(var + 1e-5f) * g[tid] + b[tid];
    out[(size_t)row * DD + tid] = (_Float16)y;
}

// ---------------------------------------------------------------------------
// WMMA GEMM:  out[M,N] = epilogue( A[M,K] @ W[N,K]^T + bias[N] )
//   4-wave workgroup computes a 64x64 tile. Per K-step the 64x32 B slice is
//   async-staged into LDS once and shared by all 4 waves; each wave reuses its
//   A fragment across 4 WMMAs (16x64 strip per wave).
//   mode 0: outH = (acc+bias)*scale            (f16)
//   mode 1: outF = residual + (acc+bias)       (f32)
//   mode 2: outH = gelu_exact(acc+bias)        (f16)
// grid = (N/64, M/64), block = (32,4)
// ---------------------------------------------------------------------------
__global__ __launch_bounds__(128)
void gemm_wmma_kernel(const _Float16* __restrict__ A,
                      const _Float16* __restrict__ W,
                      const float* __restrict__ bias,
                      const float* __restrict__ residual,
                      _Float16* __restrict__ outH,
                      float* __restrict__ outF,
                      int M, int N, int K, float scale, int mode) {
    __shared__ _Float16 ldsB[64 * 32];   // [col within tile][k within step]

    const int lane = threadIdx.x;
    const int wv = threadIdx.y;          // wave id 0..3
    const int g = lane >> 4;             // half-wave group
    const int ln = lane & 15;
    const int tn0 = blockIdx.x * 64;
    const int tm = blockIdx.y * 64 + wv * 16;
    const int t = wv * 32 + lane;        // flat thread 0..127

    v8f acc[4] = {{}, {}, {}, {}};
    const int arow = tm + ln;
    const _Float16* arowp = A + (size_t)arow * K;

    for (int kt = 0; kt < K; kt += 32) {
        // ---- async-stage B tile (64 cols x 32 k) into LDS, 2 x 16B per thread
        {
            const int q0 = 2 * t;
#pragma unroll
            for (int j = 0; j < 2; ++j) {
                const int q = q0 + j;            // 0..255 chunks of 8 halves
                const int row = q >> 2;          // col within tile
                const int part = q & 3;          // 8-half chunk within k slice
                copy16_g2l(&ldsB[row * 32 + part * 8],
                           W + (size_t)(tn0 + row) * K + kt + part * 8);
            }
        }
        // ---- A fragment (16x32, A layout) straight from global
        Frag16 a;
        {
            const _Float16* ab = arowp + kt;
            a.u[0] = *(const uint4*)(ab + 16 * g);
            a.u[1] = *(const uint4*)(ab + 16 * g + 8);
        }
        async_copy_wait();
        __syncthreads();
        // ---- 4 WMMAs reusing A across the 64-wide N strip
#pragma unroll
        for (int c = 0; c < 4; ++c) {
            Frag16 b;
            const _Float16* bb = &ldsB[(c * 16 + ln) * 32 + 16 * g];
            b.u[0] = ((const uint4*)bb)[0];
            b.u[1] = ((const uint4*)bb)[1];
            acc[c] = wmma_f16(a, b, acc[c]);
        }
        __syncthreads();
    }

#pragma unroll
    for (int c = 0; c < 4; ++c) {
        const int col = tn0 + c * 16 + ln;
        const float bv = bias[col];
#pragma unroll
        for (int r = 0; r < 8; ++r) {
            const int row = tm + r + 8 * g;      // C/D layout: row = r + 8*(lane/16)
            float val = acc[c][r] + bv;
            const size_t idx = (size_t)row * N + col;
            if (mode == 1) {
                outF[idx] = residual[idx] + val;
            } else if (mode == 2) {
                float ge = 0.5f * val * (1.0f + erff(val * 0.70710678118654752f));
                outH[idx] = (_Float16)ge;
            } else {
                outH[idx] = (_Float16)(val * scale);
            }
        }
    }
}

// ---------------------------------------------------------------------------
// Flash attention with adjacency mask + edge bias.
// 4-wave workgroup: wave w handles q-tile (blockIdx.x*4 + w) of head (b,h).
// Per 32-key step the V tile is async-staged into LDS once and shared by all
// 4 waves; scores via 2 WMMAs/wave (Q pre-scaled by 1/sqrt(HD)), online
// softmax in C-layout registers, P restaged through per-wave LDS into
// A-layout, attn@V via 2 WMMAs into two persistent f32 accumulators.
// ---------------------------------------------------------------------------
__global__ __launch_bounds__(128)
void attn_kernel(const _Float16* __restrict__ Qh,
                 const _Float16* __restrict__ Kh,
                 const _Float16* __restrict__ Vh,
                 const unsigned char* __restrict__ adj,
                 const float* __restrict__ bias2d,
                 _Float16* __restrict__ ctx) {
    __shared__ _Float16 ldsV[32 * 32];        // shared V tile (keys x feats)
    __shared__ _Float16 ldsP[4][16 * 32];     // per-wave P scratch

    const int lane = threadIdx.x;
    const int wv = threadIdx.y;
    const int g = lane >> 4;
    const int ln = lane & 15;
    const int qt = blockIdx.x * 4 + wv;       // q tile (16 rows)
    const int h  = blockIdx.y;
    const int b  = blockIdx.z;
    const int t  = wv * 32 + lane;            // flat thread 0..127

    // Q fragment (A layout), covers full head dim (K=32)
    Frag16 qa;
    {
        const int qrow = qt * 16 + ln;
        const _Float16* qb = Qh + ((size_t)(b * SS + qrow) * DD) + h * HDIM;
        qa.u[0] = *(const uint4*)(qb + 16 * g);
        qa.u[1] = *(const uint4*)(qb + 16 * g + 8);
    }

    float mrun[8], lrun[8];
    v8f o0 = {}, o1 = {};
#pragma unroll
    for (int r = 0; r < 8; ++r) { mrun[r] = -1e30f; lrun[r] = 0.0f; }

    for (int kb = 0; kb < SS / 32; ++kb) {
        // ---- async-stage shared V tile (32 keys x 32 feats), 16B per thread
        {
            const int row = t >> 2;           // key within tile
            const int part = t & 3;           // 8-half chunk within feats
            copy16_g2l(&ldsV[row * 32 + part * 8],
                       Vh + ((size_t)(b * SS + kb * 32 + row) * DD) + h * HDIM + part * 8);
        }
        // ---- K fragments: keys 0-15 and 16-31 of this tile (B layout) ----
        Frag16 kf0, kf1;
        {
            const _Float16* k0 = Kh + ((size_t)(b * SS + kb * 32 + ln) * DD) + h * HDIM + 16 * g;
            kf0.u[0] = ((const uint4*)k0)[0];
            kf0.u[1] = ((const uint4*)k0)[1];
            const _Float16* k1 = k0 + (size_t)16 * DD;
            kf1.u[0] = ((const uint4*)k1)[0];
            kf1.u[1] = ((const uint4*)k1)[1];
        }
        v8f s0 = {}, s1 = {};
        s0 = wmma_f16(qa, kf0, s0);
        s1 = wmma_f16(qa, kf1, s1);

        // ---- mask + edge bias + online softmax (per row = r + 8*g) ----
#pragma unroll
        for (int r = 0; r < 8; ++r) {
            const int m = r + 8 * g;
            const int qrow = qt * 16 + m;
            const int keyA = kb * 32 + ln;
            const int keyB = keyA + 16;
            const size_t mrow = (size_t)(b * SS + qrow) * SS;
            const size_t brow = (size_t)qrow * SS;
            float sA = adj[mrow + keyA] ? (s0[r] + bias2d[brow + keyA]) : -1e9f;
            float sB = adj[mrow + keyB] ? (s1[r] + bias2d[brow + keyB]) : -1e9f;
            // row max across the 16 lanes of this half-wave
            float tmax = fmaxf(sA, sB);
            tmax = fmaxf(tmax, __shfl_xor(tmax, 1, 32));
            tmax = fmaxf(tmax, __shfl_xor(tmax, 2, 32));
            tmax = fmaxf(tmax, __shfl_xor(tmax, 4, 32));
            tmax = fmaxf(tmax, __shfl_xor(tmax, 8, 32));
            const float mn = fmaxf(mrun[r], tmax);
            const float alpha = __expf(mrun[r] - mn);
            const float pA = __expf(sA - mn);
            const float pB = __expf(sB - mn);
            float rs = pA + pB;
            rs += __shfl_xor(rs, 1, 32);
            rs += __shfl_xor(rs, 2, 32);
            rs += __shfl_xor(rs, 4, 32);
            rs += __shfl_xor(rs, 8, 32);
            lrun[r] = lrun[r] * alpha + rs;
            mrun[r] = mn;
            o0[r] *= alpha;
            o1[r] *= alpha;
            ldsP[wv][m * 32 + ln]      = (_Float16)pA;
            ldsP[wv][m * 32 + 16 + ln] = (_Float16)pB;
        }
        async_copy_wait();
        __syncthreads();

        // ---- P (A layout) and V (B layout) from LDS; accumulate O += P @ V
        Frag16 pa;
        pa.u[0] = *(const uint4*)&ldsP[wv][ln * 32 + 16 * g];
        pa.u[1] = *(const uint4*)&ldsP[wv][ln * 32 + 16 * g + 8];
        Frag16 vf0, vf1;
#pragma unroll
        for (int e = 0; e < 16; ++e) {
            const int k = 16 * g + e;
            vf0.h[e] = ldsV[k * 32 + ln];
            vf1.h[e] = ldsV[k * 32 + 16 + ln];
        }
        o0 = wmma_f16(pa, vf0, o0);
        o1 = wmma_f16(pa, vf1, o1);
        __syncthreads();
    }

    // ---- normalize and write ctx (row-major [B*S, D], head columns) ----
#pragma unroll
    for (int r = 0; r < 8; ++r) {
        const int m = r + 8 * g;
        const int qrow = qt * 16 + m;
        const float inv = 1.0f / lrun[r];
        const size_t obase = ((size_t)(b * SS + qrow) * DD) + h * HDIM;
        ctx[obase + ln]      = (_Float16)(o0[r] * inv);
        ctx[obase + 16 + ln] = (_Float16)(o1[r] * inv);
    }
}

// ---------------------------------------------------------------------------
// Host-side launcher
// ---------------------------------------------------------------------------
extern "C" void kernel_launch(void* const* d_in, const int* in_sizes, int n_in,
                              void* d_out, int out_size, void* d_ws, size_t ws_size,
                              hipStream_t stream) {
    const float* x         = (const float*)d_in[0];
    const unsigned char* adj = (const unsigned char*)d_in[1];
    const float* edge_attr = (const float*)d_in[2];
    const int*   edge_index= (const int*)d_in[3];
    const float* ln1_g = (const float*)d_in[4];
    const float* ln1_b = (const float*)d_in[5];
    const float* wq = (const float*)d_in[6];
    const float* bq = (const float*)d_in[7];
    const float* wk = (const float*)d_in[8];
    const float* bk = (const float*)d_in[9];
    const float* wv = (const float*)d_in[10];
    const float* bv = (const float*)d_in[11];
    const float* wo = (const float*)d_in[12];
    const float* bo = (const float*)d_in[13];
    const float* ln2_g = (const float*)d_in[14];
    const float* ln2_b = (const float*)d_in[15];
    const float* w1 = (const float*)d_in[16];
    const float* b1 = (const float*)d_in[17];
    const float* w2 = (const float*)d_in[18];
    const float* b2 = (const float*)d_in[19];
    const int E = in_sizes[2];

    // ---- carve workspace ----
    char* w = (char*)d_ws;
    float*    bias2d = (float*)w;      w += (size_t)SS * SS * sizeof(float);     // 16 MB
    _Float16* nx_h   = (_Float16*)w;   w += (size_t)NROWS * DD * 2;              // 4 MB
    _Float16* q_h    = (_Float16*)w;   w += (size_t)NROWS * DD * 2;
    _Float16* k_h    = (_Float16*)w;   w += (size_t)NROWS * DD * 2;
    _Float16* v_h    = (_Float16*)w;   w += (size_t)NROWS * DD * 2;
    _Float16* ctx_h  = (_Float16*)w;   w += (size_t)NROWS * DD * 2;
    float*    x2_f   = (float*)w;      w += (size_t)NROWS * DD * sizeof(float);  // 8 MB
    _Float16* n2_h   = (_Float16*)w;   w += (size_t)NROWS * DD * 2;
    _Float16* h1_h   = (_Float16*)w;   w += (size_t)NROWS * DFF * 2;             // 16 MB
    _Float16* wq_h   = (_Float16*)w;   w += (size_t)DD * DD * 2;
    _Float16* wk_h   = (_Float16*)w;   w += (size_t)DD * DD * 2;
    _Float16* wv_h   = (_Float16*)w;   w += (size_t)DD * DD * 2;
    _Float16* wo_h   = (_Float16*)w;   w += (size_t)DD * DD * 2;
    _Float16* w1_h   = (_Float16*)w;   w += (size_t)DFF * DD * 2;
    _Float16* w2_h   = (_Float16*)w;   w += (size_t)DD * DFF * 2;
    (void)ws_size; (void)n_in; (void)out_size;

    // ---- edge bias table ----
    {
        long long n = (long long)SS * SS;
        fill_zero_f32<<<(unsigned)((n + 255) / 256), 256, 0, stream>>>(bias2d, n);
        scatter_bias_kernel<<<(E + 255) / 256, 256, 0, stream>>>(edge_attr, edge_index, bias2d, E);
    }
    // ---- weight conversion to f16 ----
    cvt_f32_to_f16<<<(DD * DD + 255) / 256, 256, 0, stream>>>(wq, wq_h, DD * DD);
    cvt_f32_to_f16<<<(DD * DD + 255) / 256, 256, 0, stream>>>(wk, wk_h, DD * DD);
    cvt_f32_to_f16<<<(DD * DD + 255) / 256, 256, 0, stream>>>(wv, wv_h, DD * DD);
    cvt_f32_to_f16<<<(DD * DD + 255) / 256, 256, 0, stream>>>(wo, wo_h, DD * DD);
    cvt_f32_to_f16<<<(DFF * DD + 255) / 256, 256, 0, stream>>>(w1, w1_h, DFF * DD);
    cvt_f32_to_f16<<<(DD * DFF + 255) / 256, 256, 0, stream>>>(w2, w2_h, DD * DFF);

    // ---- LN1 ----
    ln_kernel<<<NROWS, DD, 0, stream>>>(x, ln1_g, ln1_b, nx_h);

    // ---- QKV projections (Q pre-scaled by 1/sqrt(HD)) ----
    const float qscale = 0.1767766952966369f; // 1/sqrt(32)
    dim3 blk(32, 4);
    dim3 gD(DD / 64, NROWS / 64);
    gemm_wmma_kernel<<<gD, blk, 0, stream>>>(nx_h, wq_h, bq, nullptr, q_h, nullptr,
                                             NROWS, DD, DD, qscale, 0);
    gemm_wmma_kernel<<<gD, blk, 0, stream>>>(nx_h, wk_h, bk, nullptr, k_h, nullptr,
                                             NROWS, DD, DD, 1.0f, 0);
    gemm_wmma_kernel<<<gD, blk, 0, stream>>>(nx_h, wv_h, bv, nullptr, v_h, nullptr,
                                             NROWS, DD, DD, 1.0f, 0);

    // ---- flash attention (4 q-tiles per block share the V tile) ----
    attn_kernel<<<dim3(SS / 64, HH, BB), blk, 0, stream>>>(q_h, k_h, v_h, adj, bias2d, ctx_h);

    // ---- output projection + residual -> x2 (f32) ----
    gemm_wmma_kernel<<<gD, blk, 0, stream>>>(ctx_h, wo_h, bo, x, nullptr, x2_f,
                                             NROWS, DD, DD, 1.0f, 1);

    // ---- LN2 ----
    ln_kernel<<<NROWS, DD, 0, stream>>>(x2_f, ln2_g, ln2_b, n2_h);

    // ---- MLP up + exact GELU -> h1 (f16) ----
    gemm_wmma_kernel<<<dim3(DFF / 64, NROWS / 64), blk, 0, stream>>>(
        n2_h, w1_h, b1, nullptr, h1_h, nullptr, NROWS, DFF, DD, 1.0f, 2);

    // ---- MLP down + residual -> final output (f32) ----
    gemm_wmma_kernel<<<gD, blk, 0, stream>>>(h1_h, w2_h, b2, x2_f, nullptr, (float*)d_out,
                                             NROWS, DD, DFF, 1.0f, 1);
}
